// PNN_49151605736165
// MI455X (gfx1250) — matrix operand order repaired and checked by reading, not compiled, not run on hardware
//
#include <hip/hip_runtime.h>

#define F_FIELDS 39
#define VOCAB    200000
#define EMB      16
#define BATCH    16384
#define KTOT     (F_FIELDS * EMB)   // 624
#define NCHUNK   20                 // K padded 624 -> 640, chunks of 32

typedef __attribute__((ext_vector_type(16))) __bf16 v16bf;
typedef __attribute__((ext_vector_type(8)))  float  v8f;

static __device__ __forceinline__ v8f wmma_bf16(v16bf a, v16bf b, v8f c) {
  // D = A(16x32 bf16) x B(32x16 bf16) + C(16x16 f32)
  return __builtin_amdgcn_wmma_f32_16x16x32_bf16(false, a, false, b, (short)0, c,
                                                 false, false);
}

// Convert A-frag 8-element half from two float4's scaled by sc.
#define CVT8(A, off, e0, e1, sc)                                              \
  A[(off) + 0] = (__bf16)(e0.x * (sc)); A[(off) + 1] = (__bf16)(e0.y * (sc)); \
  A[(off) + 2] = (__bf16)(e0.z * (sc)); A[(off) + 3] = (__bf16)(e0.w * (sc)); \
  A[(off) + 4] = (__bf16)(e1.x * (sc)); A[(off) + 5] = (__bf16)(e1.y * (sc)); \
  A[(off) + 6] = (__bf16)(e1.z * (sc)); A[(off) + 7] = (__bf16)(e1.w * (sc));

#define ST4(ptr, A, j0)                                                       \
  { float4 t; t.x = A[j0]; t.y = A[(j0) + 1]; t.z = A[(j0) + 2];              \
    t.w = A[(j0) + 3]; *(float4*)(ptr) = t; }

#define LD4ADD(ptr, A, j0)                                                    \
  { float4 t = *(const float4*)(ptr); A[j0] += t.x; A[(j0) + 1] += t.y;       \
    A[(j0) + 2] += t.z; A[(j0) + 3] += t.w; }

// ---------------------------------------------------------------------------
// Pre-pass: convert W1/Wi (32x624) and lin1/lin2 (32x32) into bf16 WMMA
// B-fragments in d_ws. Slot = 32 lanes x 16 bf16 = 1KB.
//   slots 0..79 : t = slot/20 (0:W1 lo, 1:W1 hi, 2:Wi lo, 3:Wi hi), c = slot%20
//   slot 80,81  : lin1_w cols 0-15 / 16-31
//   slot 82,83  : lin2_w cols 0-15 / 16-31
// B-frag element i of lane l: K = (l>>4)*8 + i + (i>=8 ? 8 : 0) (chunk-local), N = l&15.
// ---------------------------------------------------------------------------
__global__ void pnn_convert_weights(const float* __restrict__ W1,
                                    const float* __restrict__ Wi,
                                    const float* __restrict__ L1,
                                    const float* __restrict__ L2,
                                    unsigned short* __restrict__ wf_raw) {
  int s    = blockIdx.x;
  int lane = threadIdx.x;
  int n    = lane & 15;
  int kh   = (lane >> 4) * 8;
  __bf16* dst = (__bf16*)wf_raw + ((size_t)s * 32 + lane) * 16;
  if (s < 80) {
    int t = s / 20, c = s % 20;
    const float* W = (t < 2) ? W1 : Wi;
    int d = (t & 1) * 16 + n;
#pragma unroll
    for (int i = 0; i < 16; ++i) {
      int k   = c * 32 + kh + i + (i >= 8 ? 8 : 0);
      float v = (k < KTOT) ? W[(size_t)d * KTOT + k] : 0.0f;
      dst[i]  = (__bf16)v;
    }
  } else {
    int m = s - 80;
    const float* L = (m < 2) ? L1 : L2;
    int d = (m & 1) * 16 + n;
#pragma unroll
    for (int i = 0; i < 16; ++i) dst[i] = (__bf16)L[d * 32 + kh + i + (i >= 8 ? 8 : 0)];
  }
}

// ---------------------------------------------------------------------------
// Main fused kernel. Block = 128 threads = 4 waves = 2 tiles of 16 batch rows.
// Waves (tile, parity): parity 0 -> K chunks 0..9, parity 1 -> chunks 10..19.
// Partials combined via LDS; parity-0 wave runs the WMMA MLP epilogue.
// ---------------------------------------------------------------------------
__global__ void __launch_bounds__(128) pnn_main(
    const int* __restrict__ Xi, const float* __restrict__ Xv,
    const float* __restrict__ tab, const unsigned short* __restrict__ wf_raw,
    const float* __restrict__ b1, const float* __restrict__ b2,
    const float* __restrict__ lw, const float* __restrict__ lb,
    float* __restrict__ out) {
  // per-tile scratch: 32 lanes x 36 floats (16B-aligned rows). Used first for
  // partial-accumulator exchange, then reused as the 16x32 MLP staging tile.
  __shared__ float cacc[2][32 * 36];

  const int lane = threadIdx.x & 31;
  const int wv   = threadIdx.x >> 5;
  const int tile = wv >> 1;
  const int par  = wv & 1;
  const int row0 = blockIdx.x * 32 + tile * 16;
  const int col  = lane & 15;
  const int hf   = lane >> 4;
  const int gr   = row0 + col;
  const int eoff = hf * 8;

  const int*   xir = Xi + (size_t)gr * F_FIELDS;
  const float* xvr = Xv + (size_t)gr * F_FIELDS;
  const v16bf* wfv = (const v16bf*)wf_raw;

  v8f acc0 = {}, acc1 = {}, acc2 = {}, acc3 = {};

  // ---- einsum half: this wave's K chunks (each chunk = 2 fields of 16) ----
  const int c0   = par ? 10 : 0;
  const int cend = par ? 19 : 10;
  for (int c = c0; c < cend; ++c) {
    int f0 = 2 * c, f1 = f0 + 1;
    int   i0 = xir[f0], i1 = xir[f1];
    float s0 = xvr[f0], s1 = xvr[f1];
    const float* p0 = tab + (((size_t)f0 * VOCAB + (size_t)i0) << 4) + eoff;
    const float* p1 = tab + (((size_t)f1 * VOCAB + (size_t)i1) << 4) + eoff;
    float4 e0 = *(const float4*)p0;
    float4 e1 = *(const float4*)(p0 + 4);
    float4 e2 = *(const float4*)p1;
    float4 e3 = *(const float4*)(p1 + 4);
    v16bf A;
    CVT8(A, 0, e0, e1, s0);
    CVT8(A, 8, e2, e3, s1);
    v16bf B0 = wfv[(0 * NCHUNK + c) * 32 + lane];
    v16bf B1 = wfv[(1 * NCHUNK + c) * 32 + lane];
    v16bf B2 = wfv[(2 * NCHUNK + c) * 32 + lane];
    v16bf B3 = wfv[(3 * NCHUNK + c) * 32 + lane];
    acc0 = wmma_bf16(A, B0, acc0);
    acc1 = wmma_bf16(A, B1, acc1);
    acc2 = wmma_bf16(A, B2, acc2);
    acc3 = wmma_bf16(A, B3, acc3);
  }
  if (par) {  // tail chunk c=19: field 38 valid, field 39 is zero padding
    int   i0 = xir[38];
    float s0 = xvr[38];
    const float* p0 = tab + (((size_t)38 * VOCAB + (size_t)i0) << 4) + eoff;
    float4 e0 = *(const float4*)p0;
    float4 e1 = *(const float4*)(p0 + 4);
    v16bf A;
    CVT8(A, 0, e0, e1, s0);
#pragma unroll
    for (int i = 8; i < 16; ++i) A[i] = (__bf16)0.0f;
    acc0 = wmma_bf16(A, wfv[(0 * NCHUNK + 19) * 32 + lane], acc0);
    acc1 = wmma_bf16(A, wfv[(1 * NCHUNK + 19) * 32 + lane], acc1);
    acc2 = wmma_bf16(A, wfv[(2 * NCHUNK + 19) * 32 + lane], acc2);
    acc3 = wmma_bf16(A, wfv[(3 * NCHUNK + 19) * 32 + lane], acc3);
  }

  // ---- combine the two K-halves through LDS ----
  float* cp = &cacc[tile][lane * 36];
  if (par) {
    ST4(cp + 0,  acc0, 0) ST4(cp + 4,  acc0, 4)
    ST4(cp + 8,  acc1, 0) ST4(cp + 12, acc1, 4)
    ST4(cp + 16, acc2, 0) ST4(cp + 20, acc2, 4)
    ST4(cp + 24, acc3, 0) ST4(cp + 28, acc3, 4)
  }
  __syncthreads();

  float* xsp = &cacc[tile][0];  // reused as 16x32 tile, row stride 36
  if (!par) {
    LD4ADD(cp + 0,  acc0, 0) LD4ADD(cp + 4,  acc0, 4)
    LD4ADD(cp + 8,  acc1, 0) LD4ADD(cp + 12, acc1, 4)
    LD4ADD(cp + 16, acc2, 0) LD4ADD(cp + 20, acc2, 4)
    LD4ADD(cp + 24, acc3, 0) LD4ADD(cp + 28, acc3, 4)
    // x = first_order + s*s, stage to LDS row-major (D-layout -> rows)
#pragma unroll
    for (int j = 0; j < 8; ++j) {
      float v0 = acc0[j] + acc2[j] * acc2[j];
      float v1 = acc1[j] + acc3[j] * acc3[j];
      xsp[(j + hf * 8) * 36 + col]      = v0;
      xsp[(j + hf * 8) * 36 + 16 + col] = v1;
    }
  }
  __syncthreads();

  v16bf Af;
  if (!par) {  // rebuild x as A-fragment (row = col, K-halves per bf16 A layout)
    const float* rp = xsp + col * 36;
    float4 u0 = *(const float4*)(rp + eoff);
    float4 u1 = *(const float4*)(rp + eoff + 4);
    float4 u2 = *(const float4*)(rp + 16 + eoff);
    float4 u3 = *(const float4*)(rp + 16 + eoff + 4);
    CVT8(Af, 0, u0, u1, 1.0f);
    CVT8(Af, 8, u2, u3, 1.0f);
  }
  __syncthreads();

  if (!par) {  // layer 1: relu(x @ lin1_w.T + b1), bias in the C operand
    v8f h1lo, h1hi;
    float blo = b1[col], bhi = b1[16 + col];
#pragma unroll
    for (int j = 0; j < 8; ++j) { h1lo[j] = blo; h1hi[j] = bhi; }
    h1lo = wmma_bf16(Af, wfv[80 * 32 + lane], h1lo);
    h1hi = wmma_bf16(Af, wfv[81 * 32 + lane], h1hi);
#pragma unroll
    for (int j = 0; j < 8; ++j) {
      h1lo[j] = fmaxf(h1lo[j], 0.0f);
      h1hi[j] = fmaxf(h1hi[j], 0.0f);
      xsp[(j + hf * 8) * 36 + col]      = h1lo[j];
      xsp[(j + hf * 8) * 36 + 16 + col] = h1hi[j];
    }
  }
  __syncthreads();

  if (!par) {
    v16bf Af2;
    {
      const float* rp = xsp + col * 36;
      float4 u0 = *(const float4*)(rp + eoff);
      float4 u1 = *(const float4*)(rp + eoff + 4);
      float4 u2 = *(const float4*)(rp + 16 + eoff);
      float4 u3 = *(const float4*)(rp + 16 + eoff + 4);
      CVT8(Af2, 0, u0, u1, 1.0f);
      CVT8(Af2, 8, u2, u3, 1.0f);
    }
    // layer 2: relu(h1 @ lin2_w.T + b2)
    v8f h2lo, h2hi;
    float blo = b2[col], bhi = b2[16 + col];
#pragma unroll
    for (int j = 0; j < 8; ++j) { h2lo[j] = blo; h2hi[j] = bhi; }
    h2lo = wmma_bf16(Af2, wfv[82 * 32 + lane], h2lo);
    h2hi = wmma_bf16(Af2, wfv[83 * 32 + lane], h2hi);

    // final 32->1 dot + butterfly reduce across the 16-lane column group
    float w0 = lw[col], w1 = lw[16 + col];
    float p[8];
#pragma unroll
    for (int j = 0; j < 8; ++j) {
      float a = fmaxf(h2lo[j], 0.0f);
      float b = fmaxf(h2hi[j], 0.0f);
      p[j] = a * w0 + b * w1;
    }
#pragma unroll
    for (int m = 1; m <= 8; m <<= 1) {
#pragma unroll
      for (int j = 0; j < 8; ++j) p[j] += __shfl_xor(p[j], m, 32);
    }
    if (col == 0) {
      float l0 = lb[0];
#pragma unroll
      for (int j = 0; j < 8; ++j) out[row0 + hf * 8 + j] = p[j] + l0;
    }
  }
}

// ---------------------------------------------------------------------------
extern "C" void kernel_launch(void* const* d_in, const int* in_sizes, int n_in,
                              void* d_out, int out_size, void* d_ws, size_t ws_size,
                              hipStream_t stream) {
  const int*   Xi  = (const int*)d_in[0];
  const float* Xv  = (const float*)d_in[1];
  const float* tab = (const float*)d_in[2];
  const float* W1  = (const float*)d_in[3];
  const float* Wi  = (const float*)d_in[4];
  const float* l1w = (const float*)d_in[5];
  const float* l1b = (const float*)d_in[6];
  const float* l2w = (const float*)d_in[7];
  const float* l2b = (const float*)d_in[8];
  const float* lsw = (const float*)d_in[9];
  const float* lsb = (const float*)d_in[10];
  float* out = (float*)d_out;
  unsigned short* wf = (unsigned short*)d_ws;  // 84 KB of bf16 fragments

  pnn_convert_weights<<<84, 32, 0, stream>>>(W1, Wi, l1w, l2w, wf);
  pnn_main<<<BATCH / 32, 128, 0, stream>>>(Xi, Xv, tab, wf, l1b, l2b, lsw, lsb, out);
}
